// StockBlock_63136019251738
// MI455X (gfx1250) — matrix operand order repaired and checked by reading, not compiled, not run on hardware
//
#include <hip/hip_runtime.h>

// ---------------------------------------------------------------------------
// Problem sizes (fixed by the reference)
// ---------------------------------------------------------------------------
static constexpr int Bsz  = 32;        // batch
static constexpr int Nst  = 2048;      // n_stock
static constexpr int Fdim = 1024;      // in_features
static constexpr int HS   = 4 * Nst;   // 8192 sorted-MLP hidden
static constexpr int HF   = 4 * Fdim;  // 4096 feature-MLP hidden
static constexpr int OUTF = 1024;      // out_features

typedef __attribute__((ext_vector_type(16))) __bf16       v16bf;
typedef __attribute__((ext_vector_type(8)))  float        v8f;
typedef __attribute__((ext_vector_type(4)))  unsigned int v4u;
typedef __attribute__((ext_vector_type(8)))  unsigned int v8u;
typedef __attribute__((ext_vector_type(8)))  int          v8i;
typedef __attribute__((ext_vector_type(4)))  int          v4i;

#define DI __device__ __forceinline__

// ---- CDNA5 feature probes ---------------------------------------------------
#if defined(__AMDGCN__)
  #if __has_builtin(__builtin_amdgcn_tensor_load_to_lds)
    #define USE_TDM 1
  #else
    #define USE_TDM 0
  #endif
  #if __has_builtin(__builtin_amdgcn_s_wait_tensorcnt)
    #define HAVE_WAIT_TENSOR 1
  #else
    #define HAVE_WAIT_TENSOR 0
  #endif
#else
  #define USE_TDM 0
  #define HAVE_WAIT_TENSOR 0
#endif

DI void wait_tensorcnt0() {
#if HAVE_WAIT_TENSOR
  __builtin_amdgcn_s_wait_tensorcnt(0);
#else
  asm volatile("s_wait_tensorcnt 0x0" ::: "memory");
#endif
}

DI unsigned short f2bf(float f) {           // round-to-nearest-even f32 -> bf16
  unsigned u = __float_as_uint(f);
  u += 0x7FFFu + ((u >> 16) & 1u);
  return (unsigned short)(u >> 16);
}
DI float bf2f(unsigned short h) { return __uint_as_float(((unsigned)h) << 16); }

// ---------------------------------------------------------------------------
// f32 -> bf16 weight conversion (n must be a multiple of 4)
// ---------------------------------------------------------------------------
__global__ __launch_bounds__(256) void cvt_bf16_kernel(
    const float* __restrict__ in, unsigned short* __restrict__ out, int n) {
  int i = (blockIdx.x * 256 + threadIdx.x) * 4;
  if (i + 3 < n) {
    const float* p = in + i;
    float a = p[0], b = p[1], c = p[2], d = p[3];
    unsigned lo = (unsigned)f2bf(a) | ((unsigned)f2bf(b) << 16);
    unsigned hi = (unsigned)f2bf(c) | ((unsigned)f2bf(d) << 16);
    unsigned* o = (unsigned*)(out + i);
    o[0] = lo; o[1] = hi;
  }
}

// ---------------------------------------------------------------------------
// x [B,N,F] f32  ->  xT [B,F,N] f32  (LDS-tiled transpose)
// ---------------------------------------------------------------------------
__global__ __launch_bounds__(256) void transpose_x_kernel(
    const float* __restrict__ x, float* __restrict__ xT) {
  __shared__ float t[32][33];
  const int b  = blockIdx.z;
  const int n0 = blockIdx.x * 32;
  const int f0 = blockIdx.y * 32;
  const int tx = threadIdx.x & 31;
  const int ty = threadIdx.x >> 5;           // 0..7
  const float* xb = x + (size_t)b * Nst * Fdim;
  for (int r = ty; r < 32; r += 8)
    t[r][tx] = xb[(size_t)(n0 + r) * Fdim + f0 + tx];
  __syncthreads();
  float* xTb = xT + (size_t)b * Fdim * Nst;
  for (int r = ty; r < 32; r += 8)
    xTb[(size_t)(f0 + r) * Nst + n0 + tx] = t[tx][r];
}

// ---------------------------------------------------------------------------
// Per-row stable bitonic argsort of 2048 f32 values in LDS.
// Emits p[j] (u16) and xs[j] = x[rank(x[j])] (bf16).
// ---------------------------------------------------------------------------
__global__ __launch_bounds__(256) void sort_rank_kernel(
    const float* __restrict__ xT, unsigned short* __restrict__ xs,
    unsigned short* __restrict__ p_out) {
  __shared__ float        s_val[Nst];
  __shared__ unsigned int s_idx[Nst];
  __shared__ float        s_orig[Nst];
  __shared__ unsigned int s_inv[Nst];

  const int row = blockIdx.x;                 // b*F + f
  const float* src = xT + (size_t)row * Nst;
  for (int i = threadIdx.x; i < Nst; i += 256) {
    float v = src[i];
    s_val[i] = v; s_orig[i] = v; s_idx[i] = (unsigned)i;
  }
  __syncthreads();

  for (int k = 2; k <= Nst; k <<= 1) {
    for (int j = k >> 1; j > 0; j >>= 1) {
      for (int i = threadIdx.x; i < Nst; i += 256) {
        int ix = i ^ j;
        if (ix > i) {
          bool up = ((i & k) == 0);
          float v0 = s_val[i], v1 = s_val[ix];
          unsigned i0 = s_idx[i], i1 = s_idx[ix];
          bool gt = (v0 > v1) || (v0 == v1 && i0 > i1);  // stable
          if (gt == up) {
            s_val[i] = v1; s_val[ix] = v0;
            s_idx[i] = i1; s_idx[ix] = i0;
          }
        }
      }
      __syncthreads();
    }
  }
  unsigned short* prow = p_out + (size_t)row * Nst;
  for (int j = threadIdx.x; j < Nst; j += 256) {
    prow[j] = (unsigned short)s_idx[j];
    s_inv[s_idx[j]] = (unsigned)j;            // inv = p^{-1} (rank)
  }
  __syncthreads();
  unsigned short* xrow = xs + (size_t)row * Nst;
  for (int j = threadIdx.x; j < Nst; j += 256)
    xrow[j] = f2bf(s_orig[s_inv[j]]);
}

// ---------------------------------------------------------------------------
// y [B,F,N] bf16 -- gather with p, then transpose -> yT [B,N,F] bf16
// ---------------------------------------------------------------------------
__global__ __launch_bounds__(256) void unpermute_transpose_kernel(
    const unsigned short* __restrict__ y, const unsigned short* __restrict__ p,
    unsigned short* __restrict__ yT) {
  __shared__ unsigned short t[32][33];
  const int b  = blockIdx.z;
  const int i0 = blockIdx.x * 32;
  const int f0 = blockIdx.y * 32;
  const int tx = threadIdx.x & 31;
  const int ty = threadIdx.x >> 5;
  for (int r = ty; r < 32; r += 8) {
    const size_t rowoff = ((size_t)b * Fdim + (f0 + r)) * Nst;
    unsigned short pi = p[rowoff + i0 + tx];
    t[r][tx] = y[rowoff + pi];
  }
  __syncthreads();
  for (int r = ty; r < 32; r += 8)
    yT[((size_t)b * Nst + (i0 + r)) * Fdim + f0 + tx] = t[tx][r];
}

// ---------------------------------------------------------------------------
// TN GEMM, bf16 in / f32 accumulate.
// Block tile 128(M) x 256(N), K-step 32, double-buffered LDS.
// 256 threads = 8 waves, wave grid 2(M) x 4(N); wave tile 64x64 = 4x4 WMMA
// tiles -> 16 ds_load_b128 per 16 v_wmma (1:1).
// Staging: TDM tensor_load_to_lds (TENSORcnt) with TDM-generated 16B row pad;
// fallback: global_load_b128 -> regs overlapped with compute -> ds_store.
// LDS rows padded to 80B -> conflict-free ds_load_b128 fragment reads.
// All dims are multiples of 128(M)/256(N)/32(K): no bounds checks.
// ---------------------------------------------------------------------------
template <bool RELU, bool HAS_SKIP, bool OUT_BF16>
__global__ __launch_bounds__(256) void gemm_bf16_wmma_kernel(
    const unsigned short* __restrict__ A,    // [M,K]     bf16 row-major
    const unsigned short* __restrict__ W,    // [Ncols,K] bf16 row-major
    const float* __restrict__ bias,          // [Ncols]   f32
    const unsigned short* __restrict__ skip, // [M,Ncols] bf16 (or nullptr)
    void* __restrict__ outp,                 // [M,Ncols] bf16 or f32
    int M, int Ncols, int K) {
  constexpr int ROWB = 80;                   // 64B data + 16B pad
  constexpr int ASZ  = 128 * ROWB;           // 10240
  constexpr int BSZ  = 256 * ROWB;           // 20480
  constexpr int BUF  = ASZ + BSZ;            // 30720
  __shared__ unsigned char lds[2 * BUF];

  const int t    = threadIdx.x;
  const int m0   = blockIdx.y * 128;
  const int n0   = blockIdx.x * 256;
  const int lane = t & 31;
  const int w    = t >> 5;
  const int wm   = (w & 1) * 64;
  const int wn   = (w >> 1) * 64;
  const int lh   = lane >> 4;                // lane half -> K sub-range
  const int l15  = lane & 15;

  v8f acc[4][4];
#pragma unroll
  for (int r = 0; r < 4; ++r)
#pragma unroll
    for (int c = 0; c < 4; ++c)
#pragma unroll
      for (int v = 0; v < 8; ++v) acc[r][c][v] = 0.0f;

  const int ktiles = K >> 5;

#if USE_TDM
  const unsigned ldsbase = (unsigned)(size_t)(&lds[0]);
  // Issue one TDM descriptor: tile = tileRows x 32 bf16 from a [rowsTot, K]
  // row-major tensor, packed into LDS with 16B pad after each 64B row.
  auto tdm_issue = [&](const unsigned short* gptr, unsigned ldsoff,
                       unsigned tileRows, unsigned rowsTot) {
    unsigned long long ga = (unsigned long long)(size_t)gptr;
    v4u g0;
    g0[0] = 1u;                                             // count=1
    g0[1] = ldsbase + ldsoff;                               // lds_addr
    g0[2] = (unsigned)ga;                                   // global_addr lo
    g0[3] = (unsigned)((ga >> 32) & 0x01FFFFFFu) | (2u << 30); // hi | type=2
    v8i g1;
    g1[0] = (int)((1u << 16) |                              // data_size = 2B
                  (1u << 20) |                              // pad_enable
                  (3u << 22) |                              // pad_interval: 16 dw
                  (3u << 25));                              // pad_amount: 4 dw
    g1[1] = (int)(((unsigned)K & 0xFFFFu) << 16);           // tensor_dim0 lo16
    g1[2] = (int)(((unsigned)K >> 16) | ((rowsTot & 0xFFFFu) << 16));
    g1[3] = (int)((rowsTot >> 16) | (32u << 16));           // dim1 hi | tile_dim0=32
    g1[4] = (int)(tileRows & 0xFFFFu);                      // tile_dim1, tile_dim2=0
    g1[5] = (int)K;                                         // dim0_stride lo32
    g1[6] = 0;                                              // stride hi, dim1_stride lo
    g1[7] = 0;
    v4i gz;  gz[0] = 0; gz[1] = 0; gz[2] = 0; gz[3] = 0;
    v8i gz8; gz8[0] = 0; gz8[1] = 0; gz8[2] = 0; gz8[3] = 0;
    gz8[4] = 0; gz8[5] = 0; gz8[6] = 0; gz8[7] = 0;
    // 6-arg (amdgpu-toolchain) form: (g0, g1, g2, g3, g4, cpol)
    __builtin_amdgcn_tensor_load_to_lds(g0, g1, gz, gz, gz8, 0);
  };
  auto stage = [&](int kt, int buf) {
    if (w == 0) {
      tdm_issue(A + (size_t)m0 * K + (size_t)kt * 32, (unsigned)(buf * BUF),
                128u, (unsigned)M);
      tdm_issue(W + (size_t)n0 * K + (size_t)kt * 32, (unsigned)(buf * BUF + ASZ),
                256u, (unsigned)Ncols);
    }
  };
  auto finish = [&](int /*buf*/) {
    if (w == 0) wait_tensorcnt0();
  };
#else
  v4u ra[2], rb[4];
  auto stage = [&](int kt, int /*buf*/) {        // global -> regs (overlap)
#pragma unroll
    for (int i = 0; i < 2; ++i) {                // A: 512 x 16B chunks
      int c = t + i * 256, row = c >> 2, q = c & 3;
      ra[i] = *(const v4u*)(A + (size_t)(m0 + row) * K + (size_t)kt * 32 + q * 8);
    }
#pragma unroll
    for (int i = 0; i < 4; ++i) {                // B: 1024 x 16B chunks
      int c = t + i * 256, row = c >> 2, q = c & 3;
      rb[i] = *(const v4u*)(W + (size_t)(n0 + row) * K + (size_t)kt * 32 + q * 8);
    }
  };
  auto finish = [&](int buf) {                   // regs -> LDS
#pragma unroll
    for (int i = 0; i < 2; ++i) {
      int c = t + i * 256, row = c >> 2, q = c & 3;
      *(v4u*)(lds + buf * BUF + row * ROWB + q * 16) = ra[i];
    }
#pragma unroll
    for (int i = 0; i < 4; ++i) {
      int c = t + i * 256, row = c >> 2, q = c & 3;
      *(v4u*)(lds + buf * BUF + ASZ + row * ROWB + q * 16) = rb[i];
    }
  };
#endif

  auto compute = [&](int buf) {
    const unsigned char* ldsA = lds + buf * BUF;
    const unsigned char* ldsB = lds + buf * BUF + ASZ;
    // B fragments: lanes 0-15: N=lane, K 0..7|16..23; lanes 16-31: K 8..15|24..31
    v16bf bfrag[4];
#pragma unroll
    for (int c = 0; c < 4; ++c) {
      const unsigned char* pb = ldsB + (wn + c * 16 + l15) * ROWB;
      v4u c0 = *(const v4u*)(pb + lh * 16);
      v4u c1 = *(const v4u*)(pb + 32 + lh * 16);
      v8u wv;
      wv[0] = c0[0]; wv[1] = c0[1]; wv[2] = c0[2]; wv[3] = c0[3];
      wv[4] = c1[0]; wv[5] = c1[1]; wv[6] = c1[2]; wv[7] = c1[3];
      bfrag[c] = __builtin_bit_cast(v16bf, wv);
    }
    // A fragments: lanes 0-15 K 0..7|16..23; lanes 16-31 K 8..15|24..31
#pragma unroll
    for (int r = 0; r < 4; ++r) {
      const unsigned char* pa = ldsA + (wm + r * 16 + l15) * ROWB;
      v4u c0 = *(const v4u*)(pa + lh * 16);
      v4u c1 = *(const v4u*)(pa + 32 + lh * 16);
      v8u wv;
      wv[0] = c0[0]; wv[1] = c0[1]; wv[2] = c0[2]; wv[3] = c0[3];
      wv[4] = c1[0]; wv[5] = c1[1]; wv[6] = c1[2]; wv[7] = c1[3];
      v16bf afrag = __builtin_bit_cast(v16bf, wv);
#pragma unroll
      for (int c = 0; c < 4; ++c)
        acc[r][c] = __builtin_amdgcn_wmma_f32_16x16x32_bf16(
            false, afrag, false, bfrag[c], (short)0, acc[r][c], false, false);
    }
  };

  // ---- double-buffered main loop ----
  stage(0, 0);
  finish(0);
  __syncthreads();
  for (int kt = 0; kt < ktiles; ++kt) {
    const bool more = (kt + 1) < ktiles;
    if (more) stage(kt + 1, (kt + 1) & 1);
    compute(kt & 1);
    if (more) finish((kt + 1) & 1);
    __syncthreads();
  }

  // ---- epilogue: lane l -> N = tile + (l&15); VGPR v -> M = tile + (l>>4)*8 + v
#pragma unroll
  for (int r = 0; r < 4; ++r)
#pragma unroll
    for (int c = 0; c < 4; ++c) {
      const int nn = n0 + wn + c * 16 + l15;
      const float bv = bias[nn];
#pragma unroll
      for (int v = 0; v < 8; ++v) {
        const int mm = m0 + wm + r * 16 + lh * 8 + v;
        const size_t o = (size_t)mm * Ncols + nn;
        float val = acc[r][c][v] + bv;
        if (HAS_SKIP) val += bf2f(skip[o]);
        if (RELU) val = val > 0.0f ? val : 0.0f;
        if (OUT_BF16) ((unsigned short*)outp)[o] = f2bf(val);
        else          ((float*)outp)[o] = val;
      }
    }
}

// ---------------------------------------------------------------------------
// Host-side orchestration
// ---------------------------------------------------------------------------
extern "C" void kernel_launch(void* const* d_in, const int* in_sizes, int n_in,
                              void* d_out, int out_size, void* d_ws, size_t ws_size,
                              hipStream_t stream) {
  (void)in_sizes; (void)n_in; (void)out_size; (void)ws_size;

  const float* x   = (const float*)d_in[0];
  const float* sw1 = (const float*)d_in[1];
  const float* sb1 = (const float*)d_in[2];
  const float* sw2 = (const float*)d_in[3];
  const float* sb2 = (const float*)d_in[4];
  const float* fw1 = (const float*)d_in[5];
  const float* fb1 = (const float*)d_in[6];
  const float* fw2 = (const float*)d_in[7];
  const float* fb2 = (const float*)d_in[8];

  unsigned char* ws = (unsigned char*)d_ws;
  const size_t MiB = (size_t)1 << 20;
  unsigned short* xs  = (unsigned short*)(ws);               // 128 MiB [32768,2048] bf16
  unsigned short* p   = (unsigned short*)(ws + 128  * MiB);  // 128 MiB [32768,2048] u16
  unsigned short* h   = (unsigned short*)(ws + 256  * MiB);  // 512 MiB [32768,8192] bf16
  float*          xT  = (float*)         (ws + 256  * MiB);  // 256 MiB, aliases h (dead after sort)
  unsigned short* y   = (unsigned short*)(ws + 768  * MiB);  // 128 MiB [32768,2048] bf16
  unsigned short* yT  = (unsigned short*)(ws + 896  * MiB);  // 128 MiB [65536,1024] bf16
  unsigned short* h2  = (unsigned short*)(ws + 1024 * MiB);  // 512 MiB [65536,4096] bf16
  unsigned short* w1b = (unsigned short*)(ws + 1536 * MiB);  // 32 MiB sw1 bf16
  unsigned short* w2b = (unsigned short*)(ws + 1568 * MiB);  // 32 MiB sw2 bf16
  unsigned short* w3b = (unsigned short*)(ws + 1600 * MiB);  // 8  MiB fw1 bf16
  unsigned short* w4b = (unsigned short*)(ws + 1608 * MiB);  // 8  MiB fw2 bf16

  // 1) weights -> bf16
  {
    int n1 = HS * Nst, n2 = Nst * HS, n3 = HF * Fdim, n4 = OUTF * HF;
    cvt_bf16_kernel<<<dim3((n1 / 4 + 255) / 256), 256, 0, stream>>>(sw1, w1b, n1);
    cvt_bf16_kernel<<<dim3((n2 / 4 + 255) / 256), 256, 0, stream>>>(sw2, w2b, n2);
    cvt_bf16_kernel<<<dim3((n3 / 4 + 255) / 256), 256, 0, stream>>>(fw1, w3b, n3);
    cvt_bf16_kernel<<<dim3((n4 / 4 + 255) / 256), 256, 0, stream>>>(fw2, w4b, n4);
  }
  // 2) x -> xT [B,F,N]
  transpose_x_kernel<<<dim3(Nst / 32, Fdim / 32, Bsz), 256, 0, stream>>>(x, xT);
  // 3) per-row argsort: xs (bf16) + p (u16)
  sort_rank_kernel<<<dim3(Bsz * Fdim), 256, 0, stream>>>(xT, xs, p);
  // 4) h = relu(xs @ sw1^T + sb1)            [32768, 8192]
  gemm_bf16_wmma_kernel<true, false, true>
      <<<dim3(HS / 256, (Bsz * Fdim) / 128), 256, 0, stream>>>(
          xs, w1b, sb1, nullptr, h, Bsz * Fdim, HS, Nst);
  // 5) y = h @ sw2^T + sb2 + xs              [32768, 2048]
  gemm_bf16_wmma_kernel<false, true, true>
      <<<dim3(Nst / 256, (Bsz * Fdim) / 128), 256, 0, stream>>>(
          h, w2b, sb2, xs, y, Bsz * Fdim, Nst, HS);
  // 6) un-permute with p, transpose -> yT [B*N, F]
  unpermute_transpose_kernel<<<dim3(Nst / 32, Fdim / 32, Bsz), 256, 0, stream>>>(y, p, yT);
  // 7) h2 = relu(yT @ fw1^T + fb1)           [65536, 4096]
  gemm_bf16_wmma_kernel<true, false, true>
      <<<dim3(HF / 256, (Bsz * Nst) / 128), 256, 0, stream>>>(
          yT, w3b, fb1, nullptr, h2, Bsz * Nst, HF, Fdim);
  // 8) out = h2 @ fw2^T + fb2                [65536, 1024] f32 -> d_out
  gemm_bf16_wmma_kernel<false, false, false>
      <<<dim3(OUTF / 256, (Bsz * Nst) / 128), 256, 0, stream>>>(
          h2, w4b, fb2, nullptr, d_out, Bsz * Nst, OUTF, HF);
}
